// Features_6305011990592
// MI455X (gfx1250) — matrix operand order repaired and checked by reading, not compiled, not run on hardware
//
#include <hip/hip_runtime.h>
#include <stdint.h>

// ---------------------------------------------------------------------------
// Fused KNN (k=5, Euclidean) + adaptive reduction for MI455X / gfx1250.
//
// d2[n,m] = |q_n|^2 + |m_m|^2 - 2<q_n,m_m>, GEMM via V_WMMA_F32_16X16X32_BF16
// on a bf16 hi/lo split (hi*hi + hi*lo + lo*hi, fp32 accum -> ~fp32 dots).
//
// v2: memory-bank tiles are staged in LDS with the Tensor Data Mover
// (tensor_load_to_lds, TENSORcnt, s_wait_tensorcnt), double-buffered and
// shared by all 8 waves of the block (8x less global traffic), with TDM LDS
// padding (row pitch 272B) so per-lane ds_load_b128 row reads are
// bank-conflict-free. Each lane owns one query column of the WMMA C tile, so
// top-5 selection stays register-resident.
// ---------------------------------------------------------------------------

typedef __bf16 bf16_t;
typedef __attribute__((ext_vector_type(16))) __bf16        v16bf;
typedef __attribute__((ext_vector_type(8)))  float         v8f;
typedef __attribute__((ext_vector_type(4)))  unsigned int  v4u;
typedef __attribute__((ext_vector_type(8)))  int           v8i;
typedef __attribute__((ext_vector_type(4)))  int           v4i;

#define D_DIM 128
#define KNN   5
#define FLT_BIG 3.402823466e+38f

#define TILE_ROWS   32
#define ROW_PITCH   272          // 256B row + 16B TDM pad -> bank shift of 4/row
#define PLANE_BYTES (TILE_ROWS * ROW_PITCH)

#if __has_builtin(__builtin_amdgcn_tensor_load_to_lds)
#define USE_TDM 1
#else
#define USE_TDM 0
#endif

union Frag16 { v16bf v; v4u u[2]; };
union Pack2  { __bf16 b[2]; unsigned int u; };

// --- prep: split fp32 rows into bf16 hi/lo planes + row squared-norms ------
__global__ __launch_bounds__(256) void prep_split_kernel(
    const float* __restrict__ X, bf16_t* __restrict__ hi,
    bf16_t* __restrict__ lo, float* __restrict__ sq, int rows) {
  int r = blockIdx.x * blockDim.x + threadIdx.x;
  if (r >= rows) return;
  const float* xr = X + (size_t)r * D_DIM;
  unsigned int* hp = (unsigned int*)(hi + (size_t)r * D_DIM);
  unsigned int* lp = (unsigned int*)(lo + (size_t)r * D_DIM);
  float s = 0.f;
#pragma unroll 8
  for (int k2 = 0; k2 < D_DIM / 2; ++k2) {
    float x0 = xr[2 * k2], x1 = xr[2 * k2 + 1];
    s = fmaf(x0, x0, s);
    s = fmaf(x1, x1, s);
    __bf16 h0 = (__bf16)x0, h1 = (__bf16)x1;                 // RNE
    __bf16 l0 = (__bf16)(x0 - (float)h0);                    // residual
    __bf16 l1 = (__bf16)(x1 - (float)h1);
    Pack2 ph; ph.b[0] = h0; ph.b[1] = h1; hp[k2] = ph.u;
    Pack2 pl; pl.b[0] = l0; pl.b[1] = l1; lp[k2] = pl.u;
  }
  sq[r] = s;
}

// --- register-resident ascending top-5 insert (keyed on d2) ----------------
__device__ __forceinline__ void insert5(float d, int id, float (&t)[KNN],
                                        int (&ix)[KNN]) {
  if (d < t[4]) {
    t[4] = d; ix[4] = id;
#pragma unroll
    for (int j = 4; j > 0; --j) {
      if (t[j] < t[j - 1]) {
        float tf = t[j]; t[j] = t[j - 1]; t[j - 1] = tf;
        int   ti = ix[j]; ix[j] = ix[j - 1]; ix[j - 1] = ti;
      }
    }
  }
}

#if USE_TDM
// --- TDM: 2-D tile load (32 rows x 128 bf16) into padded LDS ---------------
// D# per CDNA5 ISA: group0 {count=1, lds_addr,
// global_addr, type=2}; group1 {data_size=1(2B), pad_enable, pad_interval=5
// (64 DW = one row), pad_amount=3 (4 DW = 16B), tensor_dim0=128,
// tensor_dim1=32, tile_dim0=128, tile_dim1=32, dim0_stride=128}.
__device__ __forceinline__ void tdm_load_tile(unsigned int lds_off,
                                              const void* gptr) {
  unsigned long long ga = (unsigned long long)(uintptr_t)gptr;
  v4u g0 = { 1u,                                   // count=1, no gather
             lds_off,                              // lds_addr
             (unsigned int)ga,                     // global_addr[31:0]
             (unsigned int)((ga >> 32) & 0x01FFFFFFu) | (2u << 30) }; // +type=2
  v8i g1 = { (int)((1u << 16) | (1u << 20) | (5u << 22) | (3u << 25)),
             (int)(128u << 16),                    // tensor_dim0[15:0]=128
             (int)(32u  << 16),                    // tensor_dim1[15:0]=32
             (int)(128u << 16),                    // tile_dim0=128
             32,                                   // tile_dim1=32
             128,                                  // tensor_dim0_stride=128
             0, 0 };
  v4i z4 = {0, 0, 0, 0};
#if defined(__clang_major__) && __clang_major__ >= 23
  v8i z8 = {0, 0, 0, 0, 0, 0, 0, 0};
  __builtin_amdgcn_tensor_load_to_lds(g0, g1, z4, z4, z8, 0);
#else
  __builtin_amdgcn_tensor_load_to_lds(g0, g1, z4, z4, 0);
#endif
}
#endif

__global__ __launch_bounds__(256) void knn_wmma_kernel(
    const bf16_t* __restrict__ mem_hi, const bf16_t* __restrict__ mem_lo,
    const bf16_t* __restrict__ q_hi,  const bf16_t* __restrict__ q_lo,
    const float* __restrict__ m2,     const float* __restrict__ q2,
    const float* __restrict__ nsb,    float* __restrict__ out,
    int Nq, int Mm) {
  __shared__ __attribute__((aligned(16))) unsigned char s_tile[2][2][PLANE_BYTES];
  __shared__ float s_d[8][16][2 * KNN];
  __shared__ int   s_i[8][16][2 * KNN];

  const int lane = threadIdx.x & 31;
  const int ln   = lane & 15;     // query within wave tile / memory row in tile
  const int hh   = lane >> 4;
  const int w    = threadIdx.x >> 5;
  const int q0   = (blockIdx.x * (blockDim.x >> 5) + w) * 16;

  // ---- B fragments (queries), resident in VGPRs for the whole kernel ------
  v16bf Bh[4], Bl[4];
  {
    const bf16_t* qrh = q_hi + (size_t)(q0 + ln) * D_DIM;
    const bf16_t* qrl = q_lo + (size_t)(q0 + ln) * D_DIM;
#pragma unroll
    for (int c = 0; c < 4; ++c) {
      const int k = c * 32 + hh * 8;       // lane<16 -> K {0..7,16..23}; lane>=16 -> {8..15,24..31}
      Frag16 fh, fl;
      fh.u[0] = *(const v4u*)(qrh + k);  fh.u[1] = *(const v4u*)(qrh + k + 16);
      fl.u[0] = *(const v4u*)(qrl + k);  fl.u[1] = *(const v4u*)(qrl + k + 16);
      Bh[c] = fh.v; Bl[c] = fl.v;
    }
  }
  const float q2v = q2[q0 + ln];

  float td[KNN] = {FLT_BIG, FLT_BIG, FLT_BIG, FLT_BIG, FLT_BIG};
  int   ti[KNN] = {0, 0, 0, 0, 0};

#if USE_TDM
  const unsigned int lds_base =
      (unsigned int)(unsigned long long)(uintptr_t)(void*)&s_tile[0][0][0];
  if (w == 0) {                                  // TDM: prime buffer 0
    tdm_load_tile(lds_base, mem_hi);
    tdm_load_tile(lds_base + PLANE_BYTES, mem_lo);
  }
#endif

  for (int m0 = 0; m0 < Mm; m0 += TILE_ROWS) {
    const int cur = (m0 >> 5) & 1;
#if USE_TDM
    if (w == 0) {
      if (m0 + TILE_ROWS < Mm) {                 // async-prefetch next tile
        const int nxt = cur ^ 1;
        const unsigned int boff = lds_base + (unsigned int)(nxt * 2 * PLANE_BYTES);
        tdm_load_tile(boff, mem_hi + (size_t)(m0 + TILE_ROWS) * D_DIM);
        tdm_load_tile(boff + PLANE_BYTES, mem_lo + (size_t)(m0 + TILE_ROWS) * D_DIM);
        __builtin_amdgcn_s_wait_tensorcnt((short)2);   // current tile's 2 done
      } else {
        __builtin_amdgcn_s_wait_tensorcnt((short)0);
      }
    }
#else
    // fallback: cooperative synchronous copy into the padded LDS tile
    for (int t = threadIdx.x; t < 2 * TILE_ROWS * 16; t += 256) {
      const int plane = t >> 9, id = t & 511, row = id >> 4, seg = id & 15;
      const bf16_t* src = (plane ? mem_lo : mem_hi) + (size_t)(m0 + row) * D_DIM;
      *(v4u*)&s_tile[cur][plane][row * ROW_PITCH + seg * 16] =
          *((const v4u*)src + seg);
    }
#endif
    __syncthreads();                             // tile visible to all waves

    v8f acc0 = {0.f, 0.f, 0.f, 0.f, 0.f, 0.f, 0.f, 0.f};
    v8f acc1 = {0.f, 0.f, 0.f, 0.f, 0.f, 0.f, 0.f, 0.f};
    const unsigned char* Ah = &s_tile[cur][0][0];
    const unsigned char* Al = &s_tile[cur][1][0];
    const int r0 = ln * ROW_PITCH;               // rows 0..15
    const int r1 = (ln + 16) * ROW_PITCH;        // rows 16..31

#pragma unroll
    for (int c = 0; c < 4; ++c) {
      // A 16-bit 16x32 layout: lane<16 -> K {0..7,16..23}; lane>=16 -> {8..15,24..31}
      const int k0 = (c * 32 + hh * 8) * 2;
      const int k1 = (c * 32 + 16 + hh * 8) * 2;
      Frag16 ah0, al0, ah1, al1;                 // ds_load_b128, conflict-free
      ah0.u[0] = *(const v4u*)(Ah + r0 + k0); ah0.u[1] = *(const v4u*)(Ah + r0 + k1);
      al0.u[0] = *(const v4u*)(Al + r0 + k0); al0.u[1] = *(const v4u*)(Al + r0 + k1);
      ah1.u[0] = *(const v4u*)(Ah + r1 + k0); ah1.u[1] = *(const v4u*)(Ah + r1 + k1);
      al1.u[0] = *(const v4u*)(Al + r1 + k0); al1.u[1] = *(const v4u*)(Al + r1 + k1);

      acc0 = __builtin_amdgcn_wmma_f32_16x16x32_bf16(false, ah0.v, false, Bh[c], (short)0, acc0, false, false);
      acc1 = __builtin_amdgcn_wmma_f32_16x16x32_bf16(false, ah1.v, false, Bh[c], (short)0, acc1, false, false);
      acc0 = __builtin_amdgcn_wmma_f32_16x16x32_bf16(false, ah0.v, false, Bl[c], (short)0, acc0, false, false);
      acc1 = __builtin_amdgcn_wmma_f32_16x16x32_bf16(false, ah1.v, false, Bl[c], (short)0, acc1, false, false);
      acc0 = __builtin_amdgcn_wmma_f32_16x16x32_bf16(false, al0.v, false, Bh[c], (short)0, acc0, false, false);
      acc1 = __builtin_amdgcn_wmma_f32_16x16x32_bf16(false, al1.v, false, Bh[c], (short)0, acc1, false, false);
    }

    // C layout: lane n holds query n; rows r -> memory (m0 [+16] + hh*8 + r)
    const v8f m2a = *(const v8f*)(m2 + m0 + hh * 8);
    const v8f m2b = *(const v8f*)(m2 + m0 + 16 + hh * 8);
    const int base0 = m0 + hh * 8;
    const int base1 = m0 + 16 + hh * 8;
#pragma unroll
    for (int r = 0; r < 8; ++r)
      insert5(fmaf(-2.f, acc0[r], q2v + m2a[r]), base0 + r, td, ti);
#pragma unroll
    for (int r = 0; r < 8; ++r)
      insert5(fmaf(-2.f, acc1[r], q2v + m2b[r]), base1 + r, td, ti);

    __syncthreads();                 // all reads done before tile is reloaded
  }

  // ---- merge lane n / n+16 sorted lists through LDS, then epilogue --------
#pragma unroll
  for (int j = 0; j < KNN; ++j) {
    s_d[w][ln][hh * KNN + j] = td[j];
    s_i[w][ln][hh * KNN + j] = ti[j];
  }
  __syncthreads();

  if (hh == 0 && q0 + ln < Nq) {
    float d[KNN]; int id[KNN];
    int a = 0, b = KNN;
#pragma unroll
    for (int j = 0; j < KNN; ++j) {              // 2-way merge, keep 5 smallest
      float da = (a < KNN) ? s_d[w][ln][a] : FLT_BIG;
      float db = s_d[w][ln][b];                  // b <= 9 here
      bool ta = (da <= db);
      d[j]  = ta ? da : db;
      id[j] = ta ? s_i[w][ln][a] : s_i[w][ln][b];
      if (ta) ++a; else ++b;
    }
    float dist[KNN], sumw = 0.f, swd = 0.f, sumd = 0.f, ns = 0.f;
#pragma unroll
    for (int j = 0; j < KNN; ++j) {
      dist[j] = sqrtf(fmaxf(d[j], 1e-12f));
      float wgt = expf(-dist[j]);                // TEMPERATURE = 1.0
      sumw += wgt;
      swd  += wgt * dist[j];
      sumd += dist[j];
      ns   += nsb[id[j]];
    }
    float wd     = swd / sumw;
    float nsm    = ns * 0.2f;
    float mean_d = sumd * 0.2f;
    float var = 0.f;
#pragma unroll
    for (int j = 0; j < KNN; ++j) {
      float e = dist[j] - mean_d;
      var = fmaf(e, e, var);
    }
    var *= 0.2f;
    float cons = sqrtf(fmaxf(var, 0.f)) / fmaxf(mean_d, 1e-6f);
    float nd   = wd / fmaxf(nsm, 1e-6f);
    out[q0 + ln] = nd * (1.f + 0.5f * cons);     // CONSISTENCY_WEIGHT = 0.5
  }
}

extern "C" void kernel_launch(void* const* d_in, const int* in_sizes, int n_in,
                              void* d_out, int out_size, void* d_ws, size_t ws_size,
                              hipStream_t stream) {
  (void)n_in; (void)out_size; (void)ws_size;
  const float* query  = (const float*)d_in[0];
  const float* memory = (const float*)d_in[1];
  const float* nsb    = (const float*)d_in[2];
  // d_in[3] is k; reference fixes k=5 -> compiled in as KNN.
  const int N = in_sizes[0] / D_DIM;
  const int M = in_sizes[1] / D_DIM;

  // Workspace layout (~16.1 MB): bf16 hi/lo planes + squared norms.
  const size_t MD = (size_t)M * D_DIM, ND = (size_t)N * D_DIM;
  bf16_t* mem_hi = (bf16_t*)d_ws;
  bf16_t* mem_lo = mem_hi + MD;
  bf16_t* q_hi   = mem_lo + MD;
  bf16_t* q_lo   = q_hi + ND;
  float*  m2     = (float*)(q_lo + ND);
  float*  q2     = m2 + M;

  prep_split_kernel<<<(M + 255) / 256, 256, 0, stream>>>(memory, mem_hi, mem_lo, m2, M);
  prep_split_kernel<<<(N + 255) / 256, 256, 0, stream>>>(query,  q_hi,   q_lo,   q2, N);

  const int waves  = (N + 15) / 16;            // one wave32 per 16 queries
  const int blocks = (waves + 7) / 8;          // 8 waves / block -> 128 queries
  knn_wmma_kernel<<<blocks, 256, 0, stream>>>(mem_hi, mem_lo, q_hi, q_lo,
                                              m2, q2, nsb, (float*)d_out, N, M);
}
